// StateSpaceModel_12893491823415
// MI455X (gfx1250) — matrix operand order repaired and checked by reading, not compiled, not run on hardware
//
#include <hip/hip_runtime.h>

// StateSpaceModel for MI455X (gfx1250, wave32, WMMA + TDM).
// Per layer:  xB = h@B^T (parallel GEMM, bf16 WMMA) ->
// serial scan s_t = s_{t-1}@A^T + xB_t (4 WGs; state ping-pongs in LDS;
// xB tiles DMA'd into LDS by the Tensor Data Mover, double-buffered,
// synchronized with s_wait_tensorcnt; dual interleaved WMMA chains) ->
// y = s@C^T + (D+1)*h fused with LayerNorm (parallel GEMM).
// bf16 operands / f32 accumulate throughout.

#define BATCHN 64
#define SEQN   4096
#define NDIM   128
#define NLAYER 4
#define MTOT   (BATCHN * SEQN)   // 262144 rows
#define EPSF   1e-5f
#define TCHUNK 4                 // time steps staged per TDM transfer
#define NCHUNK (SEQN / TCHUNK)

typedef __attribute__((ext_vector_type(16))) __bf16 v16bf;
typedef __attribute__((ext_vector_type(8)))  __bf16 v8bf;
typedef __attribute__((ext_vector_type(8)))  float  v8f;
typedef __attribute__((ext_vector_type(4)))  float  v4f;
typedef __attribute__((ext_vector_type(4)))  unsigned int v4u;
typedef __attribute__((ext_vector_type(8)))  int    v8i;
typedef __attribute__((ext_vector_type(4)))  int    v4i;

static __device__ __forceinline__ v8f wmma_bf16(v16bf a, v16bf b, v8f c) {
  // D = A(16x32) * B(32x16) + C, f32 accumulate
  return __builtin_amdgcn_wmma_f32_16x16x32_bf16(false, a, false, b, (short)0, c,
                                                 false, false);
}

// A-operand (16x32, bf16) from row-major [rows x ld] bf16 matrix.
// Lane l: row = l&15; K chunks [k0+off, +8) and [k0+16+off, +8), off = (l&16)?8:0.
static __device__ __forceinline__ v16bf load_a_bf16(const __bf16* base, int ld,
                                                    int row, int k0, int lane) {
  const int off = (lane & 16) ? 8 : 0;
  const __bf16* p = base + row * ld + k0 + off;
  v8bf lo = *(const v8bf*)(p);
  v8bf hi = *(const v8bf*)(p + 16);
  v16bf r;
#pragma unroll
  for (int i = 0; i < 8; ++i) { r[i] = lo[i]; r[i + 8] = hi[i]; }
  return r;
}

// A-operand from f32 source, converting to bf16 on the fly.
static __device__ __forceinline__ v16bf load_a_f32(const float* base, int ld,
                                                   int row, int k0, int lane) {
  const int off = (lane & 16) ? 8 : 0;
  const float* p = base + row * ld + k0 + off;
  v4f x0 = *(const v4f*)(p);
  v4f x1 = *(const v4f*)(p + 4);
  v4f x2 = *(const v4f*)(p + 16);
  v4f x3 = *(const v4f*)(p + 20);
  v16bf r;
#pragma unroll
  for (int i = 0; i < 4; ++i) {
    r[i]      = (__bf16)x0[i];
    r[4 + i]  = (__bf16)x1[i];
    r[8 + i]  = (__bf16)x2[i];
    r[12 + i] = (__bf16)x3[i];
  }
  return r;
}

// B-operand (32x16, bf16) holding W[n][k] for y = x @ W^T, W row-major (N x K).
// Lane l: row n = l&15; K chunks [k0+off, +8) and [k0+16+off, +8), off = (l&16)?8:0.
static __device__ __forceinline__ v16bf load_b_f32(const float* base, int ld,
                                                   int row, int k0, int lane) {
  const float* p = base + row * ld + k0 + ((lane & 16) ? 8 : 0);
  v4f x0 = *(const v4f*)(p);
  v4f x1 = *(const v4f*)(p + 4);
  v4f x2 = *(const v4f*)(p + 16);
  v4f x3 = *(const v4f*)(p + 20);
  v16bf r;
#pragma unroll
  for (int i = 0; i < 4; ++i) {
    r[i]      = (__bf16)x0[i];
    r[4 + i]  = (__bf16)x1[i];
    r[8 + i]  = (__bf16)x2[i];
    r[12 + i] = (__bf16)x3[i];
  }
  return r;
}

// ---- TDM: DMA one 128(n) x TCHUNK(t) x 16(b) f32 tile of xB into LDS ----
// D# per CDNA5 ISA (normal mode, 3D tile):
//   group0: count=1 | lds_addr | global_addr[56:0] | type=2
//   group1: data_size=4B; tensor_dim0=128, tensor_dim1=4096;
//           tile_dim0=128, tile_dim1=TCHUNK, tile_dim2=16;
//           tensor_dim0_stride=128, tensor_dim1_stride=SEQN*128 (elements)
//   group2: tensor_dim2=64 (rest 0)    group3: 0
static __device__ __forceinline__ void tdm_load_tile(unsigned long long gaddr,
                                                     unsigned lds_addr) {
  v4u g0;
  g0[0] = 1u;                                            // count=1
  g0[1] = lds_addr;                                      // LDS byte address
  g0[2] = (unsigned)(gaddr & 0xFFFFFFFFull);             // global_addr[31:0]
  g0[3] = (unsigned)((gaddr >> 32) & 0x01FFFFFFull) | (2u << 30);  // [56:32]|type=2
  v8i g1;
  g1[0] = (int)(2u << 16);                               // data_size = 4 bytes
  g1[1] = (int)(((unsigned)NDIM) << 16);                 // tensor_dim0[15:0]=128
  g1[2] = (int)(((unsigned)SEQN) << 16);                 // tensor_dim1[15:0]=4096
  g1[3] = (int)(((unsigned)NDIM) << 16);                 // tile_dim0=128
  g1[4] = (int)((16u << 16) | (unsigned)TCHUNK);         // tile_dim1=TCHUNK, tile_dim2=16
  g1[5] = NDIM;                                          // tensor_dim0_stride lo32
  g1[6] = 0;                                             // strides mid bits (0)
  g1[7] = (int)(((unsigned)(SEQN * NDIM)) >> 16);        // tensor_dim1_stride[47:16]
  v4i g2;
  g2[0] = BATCHN;                                        // tensor_dim2 = 64
  g2[1] = 0; g2[2] = 0; g2[3] = 0;
  v4i g3 = {0, 0, 0, 0};
#if defined(__clang_major__) && (__clang_major__ >= 23)
  v8i g4 = {0, 0, 0, 0, 0, 0, 0, 0};
  __builtin_amdgcn_tensor_load_to_lds(g0, g1, g2, g3, g4, 0);
#else
  __builtin_amdgcn_tensor_load_to_lds(g0, g1, g2, g3, 0);
#endif
}

// ---------------- Kernel 1: h = bf16(x @ W_in^T + b_in) ----------------
__global__ __launch_bounds__(256) void k_gemm_in(const float* __restrict__ x,
                                                 const float* __restrict__ W,
                                                 const float* __restrict__ bias,
                                                 __bf16* __restrict__ hb) {
  const int lane = threadIdx.x & 31, wave = threadIdx.x >> 5;
  const int m0 = blockIdx.x * 16, n0 = wave * 16;
  const int arow = m0 + (lane & 15);
  const int brow = n0 + (lane & 15);
  const int nn = lane & 15, mb = (lane & 16) ? 8 : 0;
  const float bv = bias[n0 + nn];
  v8f acc;
#pragma unroll
  for (int v = 0; v < 8; ++v) acc[v] = bv;
#pragma unroll
  for (int kt = 0; kt < 4; ++kt) {
    v16bf a = load_a_f32(x, NDIM, arow, kt * 32, lane);
    v16bf b = load_b_f32(W, NDIM, brow, kt * 32, lane);
    acc = wmma_bf16(a, b, acc);
  }
#pragma unroll
  for (int v = 0; v < 8; ++v)
    hb[(m0 + v + mb) * NDIM + n0 + nn] = (__bf16)acc[v];
}

// ---------------- Kernel 2: xB = f32(h @ B_l^T) ----------------
__global__ __launch_bounds__(256) void k_gemm_xB(const __bf16* __restrict__ hb,
                                                 const float* __restrict__ Bl,
                                                 float* __restrict__ xB) {
  const int lane = threadIdx.x & 31, wave = threadIdx.x >> 5;
  const int m0 = blockIdx.x * 16, n0 = wave * 16;
  const int arow = m0 + (lane & 15);
  const int brow = n0 + (lane & 15);
  const int nn = lane & 15, mb = (lane & 16) ? 8 : 0;
  v8f acc = {};
#pragma unroll
  for (int kt = 0; kt < 4; ++kt) {
    v16bf a = load_a_bf16(hb, NDIM, arow, kt * 32, lane);
    v16bf b = load_b_f32(Bl, NDIM, brow, kt * 32, lane);
    acc = wmma_bf16(a, b, acc);
  }
#pragma unroll
  for (int v = 0; v < 8; ++v)
    xB[(m0 + v + mb) * NDIM + n0 + nn] = acc[v];
}

// ---------------- Kernel 3: serial scan s_t = s_{t-1}@A^T + xB_t ----------------
// 4 workgroups (16-batch tiles) x 8 waves (16-wide N tiles). State ping-pongs
// through LDS in bf16; A_l preloaded as B-operand registers; one barrier/step.
// xB seeds are staged into LDS by the Tensor Data Mover (double-buffered
// TCHUNK-step tiles, wave 0 issues + s_wait_tensorcnt, barrier publishes).
// Two independent WMMA accumulator chains halve the D->C RAW latency.
__global__ __launch_bounds__(256) void k_scan(const float* __restrict__ xB,
                                              const float* __restrict__ Al,
                                              __bf16* __restrict__ sout) {
  __shared__ __align__(16) __bf16 sbuf[2][16][NDIM];                //  8 KB
  __shared__ __align__(16) float  stage[2][16][TCHUNK][NDIM];       // 64 KB
  const int lane = threadIdx.x & 31, wave = threadIdx.x >> 5;
  const int b0 = blockIdx.x * 16, n0 = wave * 16;
  const int nn = lane & 15, mb = (lane & 16) ? 8 : 0;
  const int brow = n0 + nn;

  const unsigned long long gbase =
      (unsigned long long)(size_t)(const void*)xB +
      (unsigned long long)((size_t)b0 * SEQN) * NDIM * 4ull;
  const unsigned stage_lds = (unsigned)(size_t)(void*)&stage[0][0][0][0];
  const unsigned stage_half = 16u * TCHUNK * NDIM * 4u;             // 32 KB

  // Kick off chunk 0 while we set up.
  if (wave == 0)
    tdm_load_tile(gbase, stage_lds);

  v16bf Areg[4];
#pragma unroll
  for (int kt = 0; kt < 4; ++kt)
    Areg[kt] = load_b_f32(Al, NDIM, brow, kt * 32, lane);

  {  // s_0 = 0
    v8bf z = {};
    *(v8bf*)(&sbuf[0][0][0] + threadIdx.x * 8) = z;
  }

  const int aoff = (lane & 16) ? 8 : 0;
  const int arow = lane & 15;
  const int crr = threadIdx.x >> 4;          // cooperative store: row
  const int ccc = (threadIdx.x & 15) * 8;    // cooperative store: 8-elem chunk
  const int sobase = (b0 + crr) * SEQN * NDIM + ccc;   // bf16 out, coalesced
  __syncthreads();

  for (int c = 0; c < NCHUNK; ++c) {
    // Issue next chunk, then wait until chunk c has landed in LDS.
    // Tensor ops from one wave complete in order, so tensorcnt<=1 after
    // issuing c+1 means chunk c is done (ISA 08_async_tensor §7.1).
    if (wave == 0) {
      if (c + 1 < NCHUNK) {
        tdm_load_tile(gbase + (unsigned long long)((c + 1) * TCHUNK) * NDIM * 4ull,
                      stage_lds + ((unsigned)(c + 1) & 1u) * stage_half);
        __builtin_amdgcn_s_wait_tensorcnt(1);
      } else {
        __builtin_amdgcn_s_wait_tensorcnt(0);
      }
    }
    __syncthreads();  // publish staged chunk c to all waves

    const float(*st)[TCHUNK][NDIM] = stage[c & 1];
#pragma unroll
    for (int tt = 0; tt < TCHUNK; ++tt) {
      const int t = c * TCHUNK + tt;
      const int cur = t & 1, nxt = cur ^ 1;
      v8f acc0;            // chain 0 seeded with xB_t (from LDS stage)
      v8f acc1 = {};       // chain 1
#pragma unroll
      for (int v = 0; v < 8; ++v) acc0[v] = st[v + mb][tt][n0 + nn];

      // Load all four A-operand k-tiles of s_{t-1} from LDS up front.
      v16bf a[4];
#pragma unroll
      for (int kt = 0; kt < 4; ++kt) {
        const __bf16* p = &sbuf[cur][arow][kt * 32 + aoff];
        v8bf lo = *(const v8bf*)p;
        v8bf hi = *(const v8bf*)(p + 16);
#pragma unroll
        for (int i = 0; i < 8; ++i) { a[kt][i] = lo[i]; a[kt][i + 8] = hi[i]; }
      }

      // Two independent accumulation chains -> interleavable WMMAs.
      acc0 = wmma_bf16(a[0], Areg[0], acc0);
      acc1 = wmma_bf16(a[1], Areg[1], acc1);
      acc0 = wmma_bf16(a[2], Areg[2], acc0);
      acc1 = wmma_bf16(a[3], Areg[3], acc1);

#pragma unroll
      for (int v = 0; v < 8; ++v)
        sbuf[nxt][v + mb][n0 + nn] = (__bf16)(acc0[v] + acc1[v]);
      __syncthreads();

      // Coalesced 16B/lane store of the new state to global (row-major bf16).
      v8bf sv = *(const v8bf*)&sbuf[nxt][crr][ccc];
      *(v8bf*)&sout[sobase + t * NDIM] = sv;
    }
  }
}

// ---------- Kernel 4: h = LN(s@C^T + (D+1)*h) with g/beta, bf16 out ----------
__global__ __launch_bounds__(256) void k_gemm_out(const __bf16* __restrict__ s,
                                                  const __bf16* __restrict__ h,
                                                  const float* __restrict__ Cl,
                                                  const float* __restrict__ Dl,
                                                  const float* __restrict__ g,
                                                  const float* __restrict__ bt,
                                                  __bf16* __restrict__ hout) {
  __shared__ float ybuf[16][NDIM];
  const int lane = threadIdx.x & 31, wave = threadIdx.x >> 5;
  const int m0 = blockIdx.x * 16, n0 = wave * 16;
  const int arow = m0 + (lane & 15);
  const int brow = n0 + (lane & 15);
  const int nn = lane & 15, mb = (lane & 16) ? 8 : 0;
  v8f acc = {};
#pragma unroll
  for (int kt = 0; kt < 4; ++kt) {
    v16bf a = load_a_bf16(s, NDIM, arow, kt * 32, lane);
    v16bf b = load_b_f32(Cl, NDIM, brow, kt * 32, lane);
    acc = wmma_bf16(a, b, acc);
  }
  const float dn = Dl[n0 + nn] + 1.0f;  // y + residual = s@C^T + (D+1)*h
#pragma unroll
  for (int v = 0; v < 8; ++v) {
    const int mrel = v + mb;
    float hv = (float)h[(m0 + mrel) * NDIM + n0 + nn];
    ybuf[mrel][n0 + nn] = acc[v] + dn * hv;
  }
  __syncthreads();
  // LayerNorm over NDIM=128: wave w owns rows 2w, 2w+1 (16 lanes per row)
  const int r = 2 * wave + (lane >> 4);
  const int c0 = (lane & 15) * 8;
  float vals[8], ssum = 0.f, ssq = 0.f;
#pragma unroll
  for (int i = 0; i < 8; ++i) {
    float vv = ybuf[r][c0 + i];
    vals[i] = vv; ssum += vv; ssq += vv * vv;
  }
#pragma unroll
  for (int m = 8; m >= 1; m >>= 1) {   // reduce within each 16-lane half
    ssum += __shfl_xor(ssum, m, 32);
    ssq  += __shfl_xor(ssq,  m, 32);
  }
  const float mu  = ssum * (1.0f / NDIM);
  const float var = ssq * (1.0f / NDIM) - mu * mu;
  const float rs  = rsqrtf(var + EPSF);
  v8bf ov;
#pragma unroll
  for (int i = 0; i < 8; ++i)
    ov[i] = (__bf16)((vals[i] - mu) * rs * g[c0 + i] + bt[c0 + i]);
  *(v8bf*)&hout[(m0 + r) * NDIM + c0] = ov;
}

// ------------- Head: hid = relu(h_last@W1^T+b1); out = hid@W2^T+b2 -------------
__global__ __launch_bounds__(256) void k_head1(const __bf16* __restrict__ hb,
                                               const float* __restrict__ W1,
                                               const float* __restrict__ b1,
                                               float* __restrict__ hid) {
  const int idx = blockIdx.x * 256 + threadIdx.x;  // 4096 = 64 rows x 64 hidden
  const int r = idx >> 6, j = idx & 63;
  const __bf16* hr = hb + (r * SEQN + (SEQN - 1)) * NDIM;
  const float* wr = W1 + j * NDIM;
  float sacc = b1[j];
#pragma unroll 8
  for (int k = 0; k < NDIM; ++k) sacc += (float)hr[k] * wr[k];
  hid[r * 64 + j] = fmaxf(sacc, 0.0f);
}

__global__ __launch_bounds__(64) void k_head2(const float* __restrict__ hid,
                                              const float* __restrict__ W2,
                                              const float* __restrict__ b2,
                                              float* __restrict__ out) {
  const int r = threadIdx.x;  // 64
  float sacc = b2[0];
#pragma unroll 8
  for (int j = 0; j < 64; ++j) sacc += hid[r * 64 + j] * W2[j];
  out[r] = sacc;
}

extern "C" void kernel_launch(void* const* d_in, const int* in_sizes, int n_in,
                              void* d_out, int out_size, void* d_ws, size_t ws_size,
                              hipStream_t stream) {
  (void)in_sizes; (void)n_in; (void)out_size; (void)ws_size;
  const float* x    = (const float*)d_in[0];
  const float* W_in = (const float*)d_in[1];
  const float* b_in = (const float*)d_in[2];
  const float* A    = (const float*)d_in[3];   // (L,N,N)
  const float* B    = (const float*)d_in[4];   // (L,N,N)
  const float* C    = (const float*)d_in[5];   // (L,N,N)
  const float* D    = (const float*)d_in[6];   // (L,N)
  const float* ln_g = (const float*)d_in[7];
  const float* ln_b = (const float*)d_in[8];
  const float* W1   = (const float*)d_in[9];
  const float* b1   = (const float*)d_in[10];
  const float* W2   = (const float*)d_in[11];
  const float* b2   = (const float*)d_in[12];
  float* out = (float*)d_out;

  // Workspace layout: hb (bf16, 64MB) | xB (f32, 128MB) | s (bf16, 64MB) | hid
  char* ws = (char*)d_ws;
  const size_t HB_BYTES = (size_t)MTOT * NDIM * sizeof(__bf16);   //  64 MB
  const size_t XB_BYTES = (size_t)MTOT * NDIM * sizeof(float);    // 128 MB
  __bf16* hb  = (__bf16*)ws;
  float*  xB  = (float*)(ws + HB_BYTES);
  __bf16* sbf = (__bf16*)(ws + HB_BYTES + XB_BYTES);
  float*  hid = (float*)(ws + HB_BYTES + XB_BYTES + HB_BYTES);

  const int NBLK = MTOT / 16;  // 16384 workgroups for the parallel GEMMs

  k_gemm_in<<<NBLK, 256, 0, stream>>>(x, W_in, b_in, hb);
  for (int l = 0; l < NLAYER; ++l) {
    const float* Al = A + (size_t)l * NDIM * NDIM;
    const float* Bl = B + (size_t)l * NDIM * NDIM;
    const float* Cl = C + (size_t)l * NDIM * NDIM;
    k_gemm_xB<<<NBLK, 256, 0, stream>>>(hb, Bl, xB);
    k_scan<<<BATCHN / 16, 256, 0, stream>>>(xB, Al, sbf);
    k_gemm_out<<<NBLK, 256, 0, stream>>>(sbf, hb, Cl, D + l * NDIM,
                                         ln_g + l * NDIM, ln_b + l * NDIM, hb);
  }
  k_head1<<<16, 256, 0, stream>>>(hb, W1, b1, hid);
  k_head2<<<1, 64, 0, stream>>>(hid, W2, b2, out);
}
